// SSA_85633057947962
// MI455X (gfx1250) — matrix operand order repaired and checked by reading, not compiled, not run on hardware
//
#include <hip/hip_runtime.h>
#include <hip/hip_bf16.h>
#include <stdint.h>

// CDNA5 (gfx1250) wave32 WMMA bf16 kernels.
typedef __attribute__((ext_vector_type(16))) __bf16 v16bf;
typedef __attribute__((ext_vector_type(8)))  float  v8f;

#define LDS_PITCH 34   // 32 data halfwords + 2 pad -> 68B row stride, conflict-free

// ---- optional gfx1250 async global->LDS path (guarded; falls back to regs) ----
#define USE_ASYNC_LDS 0
#if defined(__has_builtin)
#if __has_builtin(__builtin_amdgcn_global_load_async_to_lds_b32) && \
    __has_builtin(__builtin_amdgcn_s_wait_asynccnt)
#undef USE_ASYNC_LDS
#define USE_ASYNC_LDS 1
#endif
#endif

#if USE_ASYNC_LDS
typedef __attribute__((address_space(1))) int* as1_iptr;   // global
typedef __attribute__((address_space(3))) int* as3_iptr;   // LDS
__device__ __forceinline__ void async_cp_b32(const unsigned* g, const unsigned short* l) {
    __builtin_amdgcn_global_load_async_to_lds_b32(
        (as1_iptr)(uintptr_t)g,
        (as3_iptr)(unsigned)(uintptr_t)l, 0, 0);
}
#endif

__device__ __forceinline__ unsigned short f2bf(float f) {
    unsigned u = __float_as_uint(f);
    u += 0x7fffu + ((u >> 16) & 1u);          // round-to-nearest-even
    return (unsigned short)(u >> 16);
}

union ABFrag { unsigned u[8]; v16bf v; };
union CDFrag { v8f v; float f[8]; };

// A fragment: 16x32 bf16; lane = (m&15)|half-sel; halves hold K {0..7,16..23}/{8..15,24..31}
__device__ __forceinline__ void load_a_frag(const unsigned short* s, int mi, int lane, unsigned f[8]) {
    const int m  = mi * 16 + (lane & 15);
    const int kb = (lane >> 4) * 8;
    const unsigned short* r = s + m * LDS_PITCH;
#pragma unroll
    for (int j = 0; j < 4; ++j) {
        f[j]     = *(const unsigned*)(r + kb + 2 * j);
        f[4 + j] = *(const unsigned*)(r + 16 + kb + 2 * j);
    }
}

// B fragment: 32x16 bf16, n-major LDS tile; halves hold K {0..7,16..23}/{8..15,24..31}
__device__ __forceinline__ void load_b_frag(const unsigned short* s, int ni, int lane, unsigned f[8]) {
    const int n  = ni * 16 + (lane & 15);
    const int kb = (lane >> 4) * 8;
    const unsigned short* r = s + n * LDS_PITCH;
#pragma unroll
    for (int j = 0; j < 4; ++j) {
        f[j]     = *(const unsigned*)(r + kb + 2 * j);
        f[4 + j] = *(const unsigned*)(r + 16 + kb + 2 * j);
    }
}

__device__ __forceinline__ v8f wmma_bf16(const unsigned a[8], const unsigned b[8], v8f c) {
    ABFrag ua, ub;
#pragma unroll
    for (int i = 0; i < 8; ++i) { ua.u[i] = a[i]; ub.u[i] = b[i]; }
    return __builtin_amdgcn_wmma_f32_16x16x32_bf16(false, ua.v, false, ub.v, (short)0, c, false, false);
}

// ---------------------------------------------------------------------------
// Unified GEMM: Y[m][n] = epilogue( sum_k A[m][k] * W[n][k] )
// Block tile 128x64, 8 waves, each wave owns a 2x2 grid of 16x16 WMMA tiles
// (4 accumulators, A/B fragments reused twice). Double-buffered LDS pipeline;
// staging via async global->LDS when available, else register staging.
// OUT_MODE 0: bf16, head-interleaved [B,H,N,dh] (q/k/v)
// OUT_MODE 1: f32, flat [M x N]                 (final projection -> d_out)
// Epilogue: relu(acc * sc[n] + sh[n])  (BN folded)
// ---------------------------------------------------------------------------
template <int OUT_MODE>
__global__ __launch_bounds__(256) void gemm_bn_relu_kernel(
    const unsigned short* __restrict__ A, const unsigned short* __restrict__ Bw,
    const float* __restrict__ sc, const float* __restrict__ sh,
    void* __restrict__ out, int M, int N, int K)
{
    __shared__ unsigned short sA[2][128 * LDS_PITCH];
    __shared__ unsigned short sB[2][64 * LDS_PITCH];
    const int tid = threadIdx.x, lane = tid & 31, w = tid >> 5;
    const int mi0 = (w & 3) * 2, ni0 = (w >> 2) * 2;
    const int m0 = blockIdx.y * 128, n0 = blockIdx.x * 64;
    const int arow = tid >> 1, acd = (tid & 1) * 8;   // A: 128 rows x 16 dwords, 8/thread
    const int brow = tid >> 2, bcd = (tid & 3) * 4;   // B:  64 rows x 16 dwords, 4/thread
    const int nk = K >> 5;
    v8f acc00 = {}, acc01 = {}, acc10 = {}, acc11 = {};

    auto compute = [&](int buf) {
        unsigned fa0[8], fa1[8], fb0[8], fb1[8];
        load_a_frag(sA[buf], mi0,     lane, fa0);
        load_a_frag(sA[buf], mi0 + 1, lane, fa1);
        load_b_frag(sB[buf], ni0,     lane, fb0);
        load_b_frag(sB[buf], ni0 + 1, lane, fb1);
        acc00 = wmma_bf16(fa0, fb0, acc00);
        acc01 = wmma_bf16(fa0, fb1, acc01);
        acc10 = wmma_bf16(fa1, fb0, acc10);
        acc11 = wmma_bf16(fa1, fb1, acc11);
    };

#if USE_ASYNC_LDS
    auto issue = [&](int ks, int buf) {
        const unsigned* gA = (const unsigned*)(A  + (size_t)(m0 + arow) * K + ks * 32);
        const unsigned* gB = (const unsigned*)(Bw + (size_t)(n0 + brow) * K + ks * 32);
        const unsigned short* lA = &sA[buf][arow * LDS_PITCH];
        const unsigned short* lB = &sB[buf][brow * LDS_PITCH];
#pragma unroll
        for (int i = 0; i < 8; ++i) async_cp_b32(gA + acd + i, lA + (acd + i) * 2);
#pragma unroll
        for (int i = 0; i < 4; ++i) async_cp_b32(gB + bcd + i, lB + (bcd + i) * 2);
    };
    issue(0, 0);
    __builtin_amdgcn_s_wait_asynccnt(0);
    __syncthreads();
    for (int i = 0; i < nk; ++i) {
        if (i + 1 < nk) issue(i + 1, (i + 1) & 1);   // DMA next tile while computing
        compute(i & 1);
        if (i + 1 < nk) __builtin_amdgcn_s_wait_asynccnt(0);
        __syncthreads();
    }
#else
    unsigned ra[8], rb[4];
    auto fetch = [&](int ks) {
        const unsigned* gA = (const unsigned*)(A  + (size_t)(m0 + arow) * K + ks * 32);
        const unsigned* gB = (const unsigned*)(Bw + (size_t)(n0 + brow) * K + ks * 32);
#pragma unroll
        for (int i = 0; i < 8; ++i) ra[i] = gA[acd + i];
#pragma unroll
        for (int i = 0; i < 4; ++i) rb[i] = gB[bcd + i];
    };
    auto stash = [&](int buf) {
        unsigned* dA = (unsigned*)&sA[buf][arow * LDS_PITCH];
        unsigned* dB = (unsigned*)&sB[buf][brow * LDS_PITCH];
#pragma unroll
        for (int i = 0; i < 8; ++i) dA[acd + i] = ra[i];
#pragma unroll
        for (int i = 0; i < 4; ++i) dB[bcd + i] = rb[i];
    };
    fetch(0);
    stash(0);
    __syncthreads();
    for (int i = 0; i < nk; ++i) {
        if (i + 1 < nk) fetch(i + 1);                // global loads overlap WMMA
        compute(i & 1);
        if (i + 1 < nk) stash((i + 1) & 1);
        __syncthreads();
    }
#endif

    CDFrag u00, u01, u10, u11;
    u00.v = acc00; u01.v = acc01; u10.v = acc10; u11.v = acc11;
    const int lm = (lane >> 4) * 8, ln = lane & 15;
    const int c0 = n0 + ni0 * 16 + ln, c1 = c0 + 16;
    const float s0 = sc[c0], h0 = sh[c0], s1 = sc[c1], h1 = sh[c1];

    auto store_pair = [&](const CDFrag& p, const CDFrag& q, int mbase) {
#pragma unroll
        for (int j = 0; j < 8; ++j) {
            const int m = mbase + j;
            const float v0 = fmaxf(p.f[j] * s0 + h0, 0.f);
            const float v1 = fmaxf(q.f[j] * s1 + h1, 0.f);
            if (OUT_MODE == 0) {
                unsigned short* ob = (unsigned short*)out;
                const int b = m >> 10, t = m & 1023;
                ob[(((size_t)(b * 8 + (c0 >> 6)) * 1024) + t) * 64 + (c0 & 63)] = f2bf(v0);
                ob[(((size_t)(b * 8 + (c1 >> 6)) * 1024) + t) * 64 + (c1 & 63)] = f2bf(v1);
            } else {
                float* of = (float*)out;
                of[(size_t)m * N + c0] = v0;
                of[(size_t)m * N + c1] = v1;
            }
        }
    };
    store_pair(u00, u01, m0 + mi0 * 16 + lm);
    store_pair(u10, u11, m0 + (mi0 + 1) * 16 + lm);
}

// ---------------------------------------------------------------------------
// Per-head T = (K^T V) * SCALE : M=64(dh), N=64(dh), K=1024(tokens).
// K,V in [B,H,N,dh] bf16; output stored TRANSPOSED (Tt[n][m]) so the next
// GEMM reads it in weight layout. grid = 64 (b*h).
// ---------------------------------------------------------------------------
__global__ __launch_bounds__(256) void ktv_kernel(
    const unsigned short* __restrict__ Kb, const unsigned short* __restrict__ Vb,
    unsigned short* __restrict__ Tt)
{
    __shared__ unsigned short sA[64 * LDS_PITCH];
    __shared__ unsigned short sB[64 * LDS_PITCH];
    const int tid = threadIdx.x, lane = tid & 31, w = tid >> 5;
    const int mi = w & 3, ni0 = (w >> 2) * 2;
    const int bh = blockIdx.x;
    const int ti = tid >> 3, cd = (tid & 7) * 4;      // 32 tokens x 32 dwords
    const size_t base = (size_t)bh * 1024 * 64;
    v8f acc0 = {}, acc1 = {};
    for (int t0 = 0; t0 < 1024; t0 += 32) {
        const unsigned* gK = (const unsigned*)(Kb + base + (size_t)(t0 + ti) * 64);
        const unsigned* gV = (const unsigned*)(Vb + base + (size_t)(t0 + ti) * 64);
#pragma unroll
        for (int i = 0; i < 4; ++i) {                 // transpose into LDS: [d][token]
            const unsigned pk = gK[cd + i], pv = gV[cd + i];
            const int d0 = (cd + i) * 2;
            sA[(d0    ) * LDS_PITCH + ti] = (unsigned short)(pk & 0xffffu);
            sA[(d0 + 1) * LDS_PITCH + ti] = (unsigned short)(pk >> 16);
            sB[(d0    ) * LDS_PITCH + ti] = (unsigned short)(pv & 0xffffu);
            sB[(d0 + 1) * LDS_PITCH + ti] = (unsigned short)(pv >> 16);
        }
        __syncthreads();
        unsigned fa[8], fb0[8], fb1[8];
        load_a_frag(sA, mi, lane, fa);
        load_b_frag(sB, ni0, lane, fb0);
        load_b_frag(sB, ni0 + 1, lane, fb1);
        acc0 = wmma_bf16(fa, fb0, acc0);
        acc1 = wmma_bf16(fa, fb1, acc1);
        __syncthreads();
    }
    CDFrag c0u, c1u; c0u.v = acc0; c1u.v = acc1;
    const int lm = (lane >> 4) * 8, ln = lane & 15;
    const int c0 = ni0 * 16 + ln, c1 = c0 + 16;
    unsigned short* tp = Tt + (size_t)bh * 4096;
#pragma unroll
    for (int j = 0; j < 8; ++j) {
        const int m = mi * 16 + lm + j;
        tp[(size_t)c0 * 64 + m] = f2bf(c0u.f[j] * 0.125f);   // fold SCALE here
        tp[(size_t)c1 * 64 + m] = f2bf(c1u.f[j] * 0.125f);
    }
}

// ---------------------------------------------------------------------------
// Per-head O = relu(Q @ T): M=1024, N=64, K=64. T read as Tt (weight layout).
// Output bf16 flat [B,N,C]. grid = (16 m-tiles, 64 bh)
// ---------------------------------------------------------------------------
__global__ __launch_bounds__(256) void qt_kernel(
    const unsigned short* __restrict__ Qb, const unsigned short* __restrict__ Tt,
    unsigned short* __restrict__ Ob)
{
    __shared__ unsigned short sA[64 * LDS_PITCH];
    __shared__ unsigned short sB[64 * LDS_PITCH];
    const int tid = threadIdx.x, lane = tid & 31, w = tid >> 5;
    const int mi = w & 3, ni0 = (w >> 2) * 2;
    const int m0 = blockIdx.x * 64;
    const int bh = blockIdx.y;
    const int row = tid >> 2, cd = (tid & 3) * 4;
    const unsigned short* Q  = Qb + (size_t)bh * 1024 * 64;
    const unsigned short* Tp = Tt + (size_t)bh * 4096;
    v8f acc0 = {}, acc1 = {};
    for (int k0 = 0; k0 < 64; k0 += 32) {
        const unsigned* gA = (const unsigned*)(Q  + (size_t)(m0 + row) * 64 + k0);
        const unsigned* gB = (const unsigned*)(Tp + (size_t)row * 64 + k0);
        unsigned* dA = (unsigned*)&sA[row * LDS_PITCH];
        unsigned* dB = (unsigned*)&sB[row * LDS_PITCH];
#pragma unroll
        for (int i = 0; i < 4; ++i) { dA[cd + i] = gA[cd + i]; dB[cd + i] = gB[cd + i]; }
        __syncthreads();
        unsigned fa[8], fb0[8], fb1[8];
        load_a_frag(sA, mi, lane, fa);
        load_b_frag(sB, ni0, lane, fb0);
        load_b_frag(sB, ni0 + 1, lane, fb1);
        acc0 = wmma_bf16(fa, fb0, acc0);
        acc1 = wmma_bf16(fa, fb1, acc1);
        __syncthreads();
    }
    CDFrag c0u, c1u; c0u.v = acc0; c1u.v = acc1;
    const int lm = (lane >> 4) * 8, ln = lane & 15;
    const int c0 = ni0 * 16 + ln, c1 = c0 + 16;
    const int b = bh >> 3, h = bh & 7;
#pragma unroll
    for (int j = 0; j < 8; ++j) {
        const int t = m0 + mi * 16 + lm + j;
        unsigned short* op = Ob + ((size_t)(b * 1024 + t)) * 512 + h * 64;
        op[c0] = f2bf(fmaxf(c0u.f[j], 0.f));   // attn_lif relu
        op[c1] = f2bf(fmaxf(c1u.f[j], 0.f));
    }
}

// ---------------------------------------------------------------------------
__global__ void cvt_bf16_kernel(const float* __restrict__ in, unsigned short* __restrict__ out, int n) {
    const int i = blockIdx.x * blockDim.x + threadIdx.x;
    if (i < n) out[i] = f2bf(in[i]);
}

__global__ void bn_coeff_kernel(const float* __restrict__ b, const float* __restrict__ g,
                                const float* __restrict__ beta, const float* __restrict__ mu,
                                const float* __restrict__ var,
                                float* __restrict__ sc, float* __restrict__ sh, int n) {
    const int i = blockIdx.x * blockDim.x + threadIdx.x;
    if (i < n) {
        const float s = g[i] * rsqrtf(var[i] + 1e-5f);
        sc[i] = s;
        sh[i] = (b[i] - mu[i]) * s + beta[i];
    }
}

// ---------------------------------------------------------------------------
extern "C" void kernel_launch(void* const* d_in, const int* in_sizes, int n_in,
                              void* d_out, int out_size, void* d_ws, size_t ws_size,
                              hipStream_t stream) {
    (void)in_sizes; (void)n_in; (void)out_size; (void)ws_size;
    const float* x = (const float*)d_in[0];

    size_t off = 0;
    auto alc = [&](size_t bytes) -> void* {
        void* p = (char*)d_ws + off;
        off += (bytes + 255) & ~(size_t)255;
        return p;
    };
    unsigned short* xb = (unsigned short*)alc(8192ull * 512 * 2);
    unsigned short* wb[4]; float* sc[4]; float* sh[4];
    for (int i = 0; i < 4; ++i) {
        wb[i] = (unsigned short*)alc(512ull * 512 * 2);
        sc[i] = (float*)alc(512 * 4);
        sh[i] = (float*)alc(512 * 4);
    }
    unsigned short* qb = (unsigned short*)alc(8192ull * 512 * 2);  // [B,H,N,dh]
    unsigned short* kb = (unsigned short*)alc(8192ull * 512 * 2);
    unsigned short* vb = (unsigned short*)alc(8192ull * 512 * 2);
    unsigned short* Tt = (unsigned short*)alc(64ull * 64 * 64 * 2);
    unsigned short* Ob = (unsigned short*)alc(8192ull * 512 * 2);  // flat [B,N,C]

    cvt_bf16_kernel<<<(8192 * 512) / 256, 256, 0, stream>>>(x, xb, 8192 * 512);
    for (int i = 0; i < 4; ++i) {
        cvt_bf16_kernel<<<(512 * 512) / 256, 256, 0, stream>>>(
            (const float*)d_in[1 + 6 * i], wb[i], 512 * 512);
        bn_coeff_kernel<<<2, 256, 0, stream>>>(
            (const float*)d_in[2 + 6 * i],  // b
            (const float*)d_in[3 + 6 * i],  // g
            (const float*)d_in[4 + 6 * i],  // beta
            (const float*)d_in[5 + 6 * i],  // mu
            (const float*)d_in[6 + 6 * i],  // var
            sc[i], sh[i], 512);
    }

    const dim3 gproj(512 / 64, 8192 / 128);
    gemm_bn_relu_kernel<0><<<gproj, 256, 0, stream>>>(xb, wb[0], sc[0], sh[0], qb, 8192, 512, 512);
    gemm_bn_relu_kernel<0><<<gproj, 256, 0, stream>>>(xb, wb[1], sc[1], sh[1], kb, 8192, 512, 512);
    gemm_bn_relu_kernel<0><<<gproj, 256, 0, stream>>>(xb, wb[2], sc[2], sh[2], vb, 8192, 512, 512);

    ktv_kernel<<<64, 256, 0, stream>>>(kb, vb, Tt);                 // T = K^T V * 0.125
    qt_kernel<<<dim3(16, 64), 256, 0, stream>>>(qb, Tt, Ob);        // O = relu(Q T)

    gemm_bn_relu_kernel<1><<<gproj, 256, 0, stream>>>(Ob, wb[3], sc[3], sh[3], d_out, 8192, 512, 512);
}
